// TernaryConv2d_31851477467322
// MI455X (gfx1250) — matrix operand order repaired and checked by reading, not compiled, not run on hardware
//
#include <hip/hip_runtime.h>
#include <hip/hip_bf16.h>
#include <stddef.h>
#include <stdint.h>

// ---------------- types ----------------
typedef _Float16 v8h  __attribute__((ext_vector_type(8)));
typedef _Float16 v16h __attribute__((ext_vector_type(16)));
typedef float    v8f  __attribute__((ext_vector_type(8)));
typedef int      v4i  __attribute__((vector_size(16)));   // matches builtin param

union V16 { v16h v; v8h h[2]; };

// ---------------- problem constants ----------------
#define B_      32
#define CIN_    256
#define COUT_   256
#define H_      56
#define W_      56
#define HP_     58          // padded
#define WP_     58
#define KK_     2304        // CIN*9, GEMM K
#define KSTEPS_ (KK_/32)    // 72
#define NPIX_   (B_*H_*W_)  // 100352, GEMM N
#define NWELEM_ (COUT_*KK_) // 589824 weights
#define THRESH_ 0.7f

// workspace layout (bytes)
#define XP_BYTES   ((size_t)B_*HP_*WP_*CIN_*2)   // 55,115,776
#define WT_BYTES   ((size_t)COUT_*KK_*2)         //  1,179,648
#define XP_OFF     ((size_t)0)
#define WT_OFF     (XP_OFF + XP_BYTES)
#define PART_OFF   (WT_OFF + WT_BYTES)
#define SUM_OFF    (PART_OFF + 256*sizeof(float))

// ---------------- CDNA5 async global->LDS copy ----------------
#if __has_builtin(__builtin_amdgcn_global_load_async_to_lds_b128)
#define ASYNC_OK 1
#else
#define ASYNC_OK 0
#endif

__device__ __forceinline__ void g2l_b128(const _Float16* g, _Float16* l) {
#if ASYNC_OK
    // GLOBAL_LOAD_ASYNC_TO_LDS_B128: per-lane 16B memory->LDS, tracked by ASYNCcnt.
    __builtin_amdgcn_global_load_async_to_lds_b128((v4i*)g, (v4i*)l, 0, 0);
#else
    *(v8h*)l = *(const v8h*)g;   // sync fallback: load->VGPR->ds_store
#endif
}

__device__ __forceinline__ void wait_async0() {
#if ASYNC_OK
#if __has_builtin(__builtin_amdgcn_s_wait_asynccnt)
    __builtin_amdgcn_s_wait_asynccnt(0);
#else
    asm volatile("s_wait_asynccnt 0x0" ::: "memory");
#endif
#endif
}

// ============================================================
// 1a. per-block partial sum of |w|  (deterministic tree reduce)
// ============================================================
__global__ __launch_bounds__(256)
void k_abs_partial(const float* __restrict__ w, float* __restrict__ partial) {
    __shared__ float sm[256];
    const int t = threadIdx.x;
    const int base = blockIdx.x * (NWELEM_ / 256);   // 2304 per block
    float s = 0.f;
    for (int i = t; i < NWELEM_ / 256; i += 256) s += fabsf(w[base + i]);
    sm[t] = s;
    __syncthreads();
    for (int o = 128; o > 0; o >>= 1) {
        if (t < o) sm[t] += sm[t + o];
        __syncthreads();
    }
    if (t == 0) partial[blockIdx.x] = sm[0];
}

// ============================================================
// 1b. final reduce of 256 partials
// ============================================================
__global__ __launch_bounds__(256)
void k_abs_final(const float* __restrict__ partial, float* __restrict__ sum) {
    __shared__ float sm[256];
    const int t = threadIdx.x;
    sm[t] = partial[t];
    __syncthreads();
    for (int o = 128; o > 0; o >>= 1) {
        if (t < o) sm[t] += sm[t + o];
        __syncthreads();
    }
    if (t == 0) *sum = sm[0];
}

// ============================================================
// 2. ternarize + fold alpha + reorder to [cout][(kh*3+kw)*256+cin] f16
// ============================================================
__global__ __launch_bounds__(256)
void k_ternize(const float* __restrict__ w, const float* __restrict__ alpha,
               const float* __restrict__ sum, _Float16* __restrict__ wt) {
    const int tid = blockIdx.x * 256 + threadIdx.x;          // 0..589823
    const float delta = THRESH_ * ((*sum) / (float)NWELEM_);
    const float a = *alpha;
    const float v = w[tid];
    float t = (v > delta) ? 1.f : ((v < -delta) ? -1.f : 0.f);
    t *= a;
    const int co  = tid / KK_;
    const int r   = tid % KK_;       // cin*9 + kh*3 + kw
    const int cin = r / 9;
    const int k9  = r % 9;           // kh*3+kw
    wt[(size_t)co * KK_ + k9 * CIN_ + cin] = (_Float16)t;
}

// ============================================================
// 3. x (NCHW f32) -> padded NHWC f16 [B][58][58][256]
//    one warp per padded pixel; lane handles 8 channels (16B store)
// ============================================================
__global__ __launch_bounds__(256)
void k_pad(const float* __restrict__ x, _Float16* __restrict__ xp) {
    const int t    = blockIdx.x * 256 + threadIdx.x;
    const int lane = t & 31;
    const int pix  = t >> 5;                    // 0 .. 32*58*58-1
    const int b    = pix / (HP_ * WP_);
    const int rem  = pix % (HP_ * WP_);
    const int ih   = rem / WP_;
    const int iw   = rem % WP_;
    const int c0   = lane * 8;

    v8h out;
    if (ih >= 1 && ih <= H_ && iw >= 1 && iw <= W_) {
        const float* src = x + (((size_t)b * CIN_ + c0) * H_ + (ih - 1)) * W_ + (iw - 1);
        #pragma unroll
        for (int i = 0; i < 8; ++i)
            out[i] = (_Float16)__builtin_nontemporal_load(&src[(size_t)i * (H_ * W_)]);
    } else {
        #pragma unroll
        for (int i = 0; i < 8; ++i) out[i] = (_Float16)0.f;
    }
    *(v8h*)&xp[(size_t)pix * CIN_ + c0] = out;
}

// ============================================================
// 4. implicit GEMM:  out[co][p] = sum_k wt[co][k] * xcol[k][p]
//    M=cout (A), N=pixels (B), K=2304, K-step 32 via v_wmma_f32_16x16x32_f16
//    block tile 128(co) x 128(pix), 8 waves, each wave: 16 co x 128 pix
//    double-buffered LDS, async global->LDS staging pipelined with WMMA
// ============================================================
struct GemmCtx {
    const _Float16* xbase;   // per-thread global staging src (X), ks-invariant part
    const _Float16* wbase;   // per-thread global staging src (W)
    _Float16* stX[2];        // per-thread LDS staging dst, buffer 0/1
    _Float16* stW[2];
    const _Float16* lXb[2];  // LDS buffer bases for fragment reads
    const _Float16* lWb[2];
    int aoff0, aoff1, boff;  // fragment offsets (f16 units)
};

__device__ __forceinline__ void prefetch_step(const GemmCtx& c, int ks, int buf) {
    const int w9   = ks >> 3;
    const int cinb = (ks & 7) * 32;
    const int kh   = w9 / 3;
    const int kw   = w9 % 3;
    const _Float16* gx = c.xbase + (kh * WP_ + kw) * CIN_ + cinb;
    const _Float16* gw = c.wbase + w9 * CIN_ + cinb;
    g2l_b128(gx,     c.stX[buf]);
    g2l_b128(gx + 8, c.stX[buf] + 8);
    g2l_b128(gw,     c.stW[buf]);
    g2l_b128(gw + 8, c.stW[buf] + 8);
}

__device__ __forceinline__ void kstep(const GemmCtx& c, int ks, int buf, v8f acc[8]) {
    wait_async0();          // my stage-ks async copies have landed in LDS
    __syncthreads();        // everyone's have
    if (ks + 1 < KSTEPS_)   // pipeline: stage ks+1 into the other buffer
        prefetch_step(c, ks + 1, buf ^ 1);

    const _Float16* lw = c.lWb[buf];
    const _Float16* lx = c.lXb[buf];

    V16 a;
    a.h[0] = *(const v8h*)(lw + c.aoff0);
    a.h[1] = *(const v8h*)(lw + c.aoff1);
    V16 bf[8];
    #pragma unroll
    for (int ns = 0; ns < 8; ++ns) {
        bf[ns].h[0] = *(const v8h*)(lx + ns * 16 * 32 + c.boff);
        bf[ns].h[1] = *(const v8h*)(lx + ns * 16 * 32 + c.boff + 16);
    }
    #pragma unroll
    for (int ns = 0; ns < 8; ++ns) {
        acc[ns] = __builtin_amdgcn_wmma_f32_16x16x32_f16(
            false, a.v, false, bf[ns].v, (short)0, acc[ns], false, false);
    }
}

__global__ __launch_bounds__(256)
void k_gemm(const _Float16* __restrict__ xp, const _Float16* __restrict__ wt,
            const float* __restrict__ bias, float* __restrict__ out) {
    __shared__ __align__(16) _Float16 lW[2][128 * 32];   // A tiles: [co_local][k]
    __shared__ __align__(16) _Float16 lX[2][128 * 32];   // B tiles: [p_local][k]

    const int tid  = threadIdx.x;
    const int wave = tid >> 5;
    const int lane = tid & 31;
    const int lo16 = lane & 15;
    const int half = lane >> 4;

    const int ptile = blockIdx.x;          // 0..783   (pixels / 128)
    const int co0   = blockIdx.y * 128;    // 0 or 128

    // ---- staging mapping: thread -> (row 0..127, 16-f16 chunk 0..1) ----
    const int srow  = tid >> 1;
    const int schnk = (tid & 1) * 16;

    // X staging source pixel
    const int p_s  = ptile * 128 + srow;
    const int b_s  = p_s / (H_ * W_);
    const int s_s  = p_s % (H_ * W_);
    const int oh_s = s_s / W_;
    const int ow_s = s_s % W_;

    GemmCtx c;
    c.xbase = xp + ((size_t)(b_s * HP_ + oh_s) * WP_ + ow_s) * CIN_ + schnk;
    c.wbase = wt + (size_t)(co0 + srow) * KK_ + schnk;
    c.stX[0] = &lX[0][srow * 32 + schnk];
    c.stX[1] = &lX[1][srow * 32 + schnk];
    c.stW[0] = &lW[0][srow * 32 + schnk];
    c.stW[1] = &lW[1][srow * 32 + schnk];
    c.lXb[0] = &lX[0][0];  c.lXb[1] = &lX[1][0];
    c.lWb[0] = &lW[0][0];  c.lWb[1] = &lW[1][0];

    // fragment LDS offsets (CDNA5 wave32 layouts)
    // A (16x32 f16): lane<16 -> K {0..7, 16..23}; lane>=16 -> K {8..15, 24..31}
    const int arow = wave * 16 + lo16;
    c.aoff0 = arow * 32 + half * 8;
    c.aoff1 = arow * 32 + 16 + half * 8;
    // B (32x16 f16): lane<16 -> K {0..7, 16..23}; lane>=16 -> K {8..15, 24..31} ; col = lane&15
    c.boff  = lo16 * 32 + half * 8;

    v8f acc[8] = {};

    prefetch_step(c, 0, 0);                 // prologue: stage ks=0 into buf0

    #pragma unroll 1
    for (int kp = 0; kp < KSTEPS_ / 2; ++kp) {   // 36 double-steps
        kstep(c, 2 * kp,     0, acc);
        kstep(c, 2 * kp + 1, 1, acc);
    }

    // ---- epilogue: C 16x16 f32 layout -> lane = col(pixel), vgpr r = row(cout)
    // nontemporal: out is streamed once; keep L2 for the re-read xp tiles
    const int coBase = co0 + wave * 16 + half * 8;
    #pragma unroll
    for (int ns = 0; ns < 8; ++ns) {
        const int p = ptile * 128 + ns * 16 + lo16;
        const int b = p / (H_ * W_);
        const int s = p % (H_ * W_);
        const size_t obase = (size_t)b * COUT_ * (H_ * W_) + s;
        #pragma unroll
        for (int r = 0; r < 8; ++r) {
            const int co = coBase + r;
            __builtin_nontemporal_store(acc[ns][r] + bias[co],
                                        &out[obase + (size_t)co * (H_ * W_)]);
        }
    }
}

// ============================================================
// launcher
// ============================================================
extern "C" void kernel_launch(void* const* d_in, const int* in_sizes, int n_in,
                              void* d_out, int out_size, void* d_ws, size_t ws_size,
                              hipStream_t stream) {
    const float* x     = (const float*)d_in[0];
    const float* w     = (const float*)d_in[1];
    const float* alpha = (const float*)d_in[2];
    const float* bias  = (const float*)d_in[3];
    float*       out   = (float*)d_out;

    char* ws = (char*)d_ws;
    _Float16* xp      = (_Float16*)(ws + XP_OFF);
    _Float16* wt      = (_Float16*)(ws + WT_OFF);
    float*    partial = (float*)(ws + PART_OFF);
    float*    sum     = (float*)(ws + SUM_OFF);

    k_abs_partial<<<256, 256, 0, stream>>>(w, partial);
    k_abs_final  <<<1,   256, 0, stream>>>(partial, sum);
    k_ternize    <<<NWELEM_ / 256, 256, 0, stream>>>(w, alpha, sum, wt);
    k_pad        <<<(B_ * HP_ * WP_ * 32) / 256, 256, 0, stream>>>(x, xp);

    dim3 grid(NPIX_ / 128, COUT_ / 128);
    k_gemm<<<grid, 256, 0, stream>>>(xp, wt, bias, out);
}
